// Sopa_32710470926801
// MI455X (gfx1250) — matrix operand order repaired and checked by reading, not compiled, not run on hardware
//
#include <hip/hip_runtime.h>
#include <hip/hip_bf16.h>

typedef __attribute__((ext_vector_type(16))) _Float16 v16h;
typedef __attribute__((ext_vector_type(8)))  float    v8f;

#define HH   256
#define DD   1200
#define LL   512
#define BB   64
#define LB   32768          /* L*B */
#define NEGV (-100.0f)

// ---------------------------------------------------------------------------
// GEMM: sc[d][n] = diags[d] . x_flat[n] + bias[d],  d<1200, n<32768, K=256
// fp32 inputs converted to f16 in-register, f32 accumulation via WMMA.
// One wave per 64x64 output block: 4x4 tiles of 16x16, K-loop step 32.
// ---------------------------------------------------------------------------
__global__ __launch_bounds__(32)
void sopa_gemm_wmma(const float* __restrict__ diags,
                    const float* __restrict__ x,     // (L*B, H) row-major
                    const float* __restrict__ bias,
                    float* __restrict__ sc)          // (D, L*B) row-major
{
    const int lane  = threadIdx.x;
    const int lhalf = lane & 15;
    const int hi    = lane >> 4;            // 0 or 1
    const int n0    = blockIdx.x * 64;      // 512 n-blocks
    const int mtb   = blockIdx.y * 4;       // 19 m-blocks (75 tiles of 16)

    v8f acc[4][4] = {};                     // 16 accumulators, f32

    for (int k0 = 0; k0 < HH; k0 += 32) {
        // --- A fragments: 16x32 f16 layout: lane=row (mod 16); hi half -> K+8;
        //     elements 0..7 -> K = k0+e, elements 8..15 -> K = k0+16+(e-8)
        v16h a[4];
        #pragma unroll
        for (int mt = 0; mt < 4; ++mt) {
            int dt  = mtb + mt;
            int dtc = dt < 75 ? dt : 74;    // clamp OOB tile (result discarded)
            const float* ap = diags + (size_t)(dtc * 16 + lhalf) * HH + k0 + hi * 8;
            #pragma unroll
            for (int e = 0; e < 8; ++e) {
                a[mt][e]     = (_Float16)ap[e];
                a[mt][8 + e] = (_Float16)ap[16 + e];
            }
        }
        // --- B fragments: 32x16 f16 layout: lane=col (mod 16); hi half -> K+8;
        //     elements 0..7 -> K = k0+e, elements 8..15 -> K = k0+16+(e-8)
        #pragma unroll
        for (int nt = 0; nt < 4; ++nt) {
            v16h bfrag;
            const float* bp = x + (size_t)(n0 + nt * 16 + lhalf) * HH + k0 + hi * 8;
            #pragma unroll
            for (int e = 0; e < 8; ++e) {
                bfrag[e]     = (_Float16)bp[e];
                bfrag[8 + e] = (_Float16)bp[16 + e];
            }
            #pragma unroll
            for (int mt = 0; mt < 4; ++mt)
                acc[mt][nt] = __builtin_amdgcn_wmma_f32_16x16x32_f16(
                    false, a[mt], false, bfrag, (short)0, acc[mt][nt],
                    false, false);
        }
    }

    // Epilogue: C/D layout: VGPR r, lane -> row = r + 8*hi, col = lane&15
    #pragma unroll
    for (int mt = 0; mt < 4; ++mt) {
        int dt = mtb + mt;
        if (dt < 75) {
            #pragma unroll
            for (int r = 0; r < 8; ++r) {
                int   row = dt * 16 + r + hi * 8;
                float bv  = bias[row];
                #pragma unroll
                for (int nt = 0; nt < 4; ++nt)
                    sc[(size_t)row * LB + n0 + nt * 16 + lhalf] = acc[mt][nt][r] + bv;
            }
        }
    }
}

// ---------------------------------------------------------------------------
// Max-plus scan over L=512. One lane per (b,p) chain; lanes of consecutive p
// read contiguous 48B slices of sc (the torch .view is the identity on the
// flat buffer since P*2*M == D). Register double-buffer + deep prefetch.
// ---------------------------------------------------------------------------
__global__ __launch_bounds__(128)
void sopa_scan(const float* __restrict__ sc,
               const int*   __restrict__ input_len,
               const float* __restrict__ epsilon,     // (P, 5)
               const float* __restrict__ slp,         // (1,)
               float* __restrict__ out_scores,        // (B, P)
               float* __restrict__ out_hidden)        // (L+1, B, P, 6)
{
    const int b = blockIdx.x;
    const int p = threadIdx.x;
    if (p >= 100) return;

    const float sl   = slp[0];
    const int   len  = input_len[b];
    const int   endi = (p < 50) ? 4 : 5;

    float eps[5];
    #pragma unroll
    for (int i = 0; i < 5; ++i) eps[i] = epsilon[p * 5 + i];

    float h[6];
    h[0] = 0.f;
    #pragma unroll
    for (int i = 1; i < 6; ++i) h[i] = NEGV;
    float s = NEGV;

    // all_hiddens[0] = h0
    {
        float* o = out_hidden + (size_t)b * 600 + p * 6;
        o[0] = 0.f;
        #pragma unroll
        for (int i = 1; i < 6; ++i) o[i] = NEGV;
    }

    const float* base = sc + (size_t)b * LL * DD + p * 12;

    float4 m0 = *(const float4*)(base + 0);
    float4 m1 = *(const float4*)(base + 4);
    float4 m2 = *(const float4*)(base + 8);

    for (int l = 0; l < LL; ++l) {
        // prefetch 8 steps ahead (global_prefetch_b8): covers our 48B slice
        if (l + 8 < LL)
            __builtin_prefetch((const void*)(base + (size_t)(l + 8) * DD), 0, 0);

        // issue next step's loads before computing (double buffer)
        const float* nb = base + (size_t)((l + 1 < LL) ? (l + 1) : l) * DD;
        float4 n0v = *(const float4*)(nb + 0);
        float4 n1v = *(const float4*)(nb + 4);
        float4 n2v = *(const float4*)(nb + 8);

        const float m[12] = { m0.x, m0.y, m0.z, m0.w,
                              m1.x, m1.y, m1.z, m1.w,
                              m2.x, m2.y, m2.z, m2.w };
        // m[0..5] = j=0 (self-loop scores), m[6..11] = j=1 (main-path scores)

        float ae[6];
        ae[0] = fmaxf(h[0], NEGV);
        #pragma unroll
        for (int i = 1; i < 6; ++i) ae[i] = fmaxf(h[i], h[i - 1] + eps[i - 1]);

        float hn[6];
        hn[0] = fmaxf(0.f, sl + ae[0] + m[0]);
        #pragma unroll
        for (int i = 1; i < 6; ++i)
            hn[i] = fmaxf(ae[i - 1] + m[6 + i - 1], sl + ae[i] + m[i]);
        #pragma unroll
        for (int i = 0; i < 6; ++i) h[i] = hn[i];

        if (len > l) s = fmaxf(s, h[endi]);

        float* o = out_hidden + (size_t)(l + 1) * (BB * 600) + (size_t)b * 600 + p * 6;
        *(float2*)(o + 0) = make_float2(h[0], h[1]);
        *(float2*)(o + 2) = make_float2(h[2], h[3]);
        *(float2*)(o + 4) = make_float2(h[4], h[5]);

        m0 = n0v; m1 = n1v; m2 = n2v;
    }

    out_scores[b * 100 + p] = s;
}

// ---------------------------------------------------------------------------
extern "C" void kernel_launch(void* const* d_in, const int* in_sizes, int n_in,
                              void* d_out, int out_size, void* d_ws, size_t ws_size,
                              hipStream_t stream) {
    const float* input      = (const float*)d_in[0];  // (512, 64, 256) fp32
    const int*   input_len  = (const int*)  d_in[1];  // (64,) int32
    const float* diags      = (const float*)d_in[2];  // (1200, 256)
    const float* bias       = (const float*)d_in[3];  // (1200, 1)
    const float* epsilon    = (const float*)d_in[4];  // (100, 5)
    const float* slp        = (const float*)d_in[5];  // (1,)

    float* sc = (float*)d_ws;                         // 1200*32768 fp32 = 157.3 MB
    float* out_scores = (float*)d_out;                // (64, 100)
    float* out_hidden = out_scores + BB * 100;        // (513, 64, 100, 6)

    dim3 g1(LB / 64, 19), b1(32);
    sopa_gemm_wmma<<<g1, b1, 0, stream>>>(diags, input, bias, sc);

    dim3 g2(BB), b2(128);
    sopa_scan<<<g2, b2, 0, stream>>>(sc, input_len, epsilon, slp,
                                     out_scores, out_hidden);
}
